// CrossAttention_10840497455807
// MI455X (gfx1250) — matrix operand order repaired and checked by reading, not compiled, not run on hardware
//
#include <hip/hip_runtime.h>
#include <hip/hip_bf16.h>
#include <math.h>

// Problem constants
#define B_  16
#define NP_ 4096
#define NT_ 77
#define NTS_ 80          // score width: NT padded to multiple of 16
#define NTP_ 96          // prob width: NT padded to multiple of 32 (WMMA K-chunks)
#define QD_ 512
#define CD_ 768
#define H_  8
#define DH_ 64
#define INNER_ 512

typedef __attribute__((ext_vector_type(16))) __bf16 v16bf;
typedef __attribute__((ext_vector_type(8)))  float  v8f;

union FragBF {
    v16bf v;
    unsigned short u[16];
    uint4 q[2];
};

__device__ __forceinline__ unsigned short f2bf(float x) {
    unsigned int u = __float_as_uint(x);
    unsigned int r = u + 0x7FFFu + ((u >> 16) & 1u);   // round-to-nearest-even
    return (unsigned short)(r >> 16);
}

// A-matrix fragment (16x32 bf16, M x K) from row-major bf16 buffer (LDS or global).
// lane l: row = row0 + (l&15); elements e: k = k0 + (e&7) + 16*(e>>3) + 8*(l>>4)
__device__ __forceinline__ FragBF ldA_bf(const unsigned short* base, int row0, int k0,
                                         int ld, int lane) {
    const int lo = lane & 15, hi = lane >> 4;
    const unsigned short* p = base + (size_t)(row0 + lo) * ld + k0 + 8 * hi;
    FragBF f;
    f.q[0] = *(const uint4*)p;          // k = k0+8*hi    .. +7
    f.q[1] = *(const uint4*)(p + 16);   // k = k0+16+8*hi .. +7
    return f;
}

// B-matrix fragment (32x16 bf16, K x N) from "transposed" storage T[n][k] (ld = K stride).
// lane l: col n = n0 + (l&15); elements e: k = k0 + (e&7) + 16*(e>>3) + 8*(l>>4)
__device__ __forceinline__ FragBF ldB_bf(const unsigned short* baseT, int n0, int k0,
                                         int ld, int lane) {
    const int lo = lane & 15, hi = lane >> 4;
    const unsigned short* p = baseT + (size_t)(n0 + lo) * ld + k0 + 8 * hi;
    FragBF f;
    f.q[0] = *(const uint4*)p;
    f.q[1] = *(const uint4*)(p + 16);
    return f;
}

// A fragment from row-major fp32 global with row guard (rows >= nrows read as zero).
__device__ __forceinline__ FragBF ldA_f32g(const float* base, int nrows, int row0, int k0,
                                           int ld, int lane) {
    const int lo = lane & 15, hi = lane >> 4;
    const int r = row0 + lo;
    FragBF f;
    if (r < nrows) {
        const float* p = base + (size_t)r * ld + k0 + 8 * hi;
        float4 a0 = ((const float4*)p)[0];
        float4 a1 = ((const float4*)p)[1];
        float4 c0 = ((const float4*)(p + 16))[0];
        float4 c1 = ((const float4*)(p + 16))[1];
        f.u[0] = f2bf(a0.x); f.u[1] = f2bf(a0.y); f.u[2]  = f2bf(a0.z); f.u[3]  = f2bf(a0.w);
        f.u[4] = f2bf(a1.x); f.u[5] = f2bf(a1.y); f.u[6]  = f2bf(a1.z); f.u[7]  = f2bf(a1.w);
        f.u[8] = f2bf(c0.x); f.u[9] = f2bf(c0.y); f.u[10] = f2bf(c0.z); f.u[11] = f2bf(c0.w);
        f.u[12] = f2bf(c1.x); f.u[13] = f2bf(c1.y); f.u[14] = f2bf(c1.z); f.u[15] = f2bf(c1.w);
    } else {
        f.q[0] = make_uint4(0, 0, 0, 0);
        f.q[1] = make_uint4(0, 0, 0, 0);
    }
    return f;
}

__device__ __forceinline__ v8f wmma_bf(const FragBF& a, const FragBF& b, v8f c) {
    return __builtin_amdgcn_wmma_f32_16x16x32_bf16(false, a.v, false, b.v,
                                                   (short)0, c, false, false);
}

// ---------------------------------------------------------------------------
// Kernel 1: fp32 weight W[k][n] -> bf16 transposed WT[n][k]
// ---------------------------------------------------------------------------
__global__ __launch_bounds__(256) void transpose_w(const float* __restrict__ src,
                                                   unsigned short* __restrict__ dst,
                                                   int K, int N) {
    int idx = blockIdx.x * 256 + threadIdx.x;
    if (idx < K * N) {
        int n = idx / K, k = idx % K;
        dst[idx] = f2bf(src[(size_t)k * N + n]);
    }
}

// ---------------------------------------------------------------------------
// Kernel 2: K/V projection.
//   K  -> kws [b][h][t(96)][d(64)]  bf16   (B-frag friendly for Q*K^T)
//   V  -> vt  [b][h][d(64)][t(96)]  bf16   (B-frag friendly for P*V)
// grid (B, 6 row-tiles of 16 t's), 256 threads (8 waves).
// Per kk: batch-load A + all 8 B-frags into distinct registers (one clause /
// one wait), then 8 back-to-back WMMAs.
// ---------------------------------------------------------------------------
__global__ __launch_bounds__(256) void kv_proj(const float* __restrict__ ctx,
                                               const unsigned short* __restrict__ wkT,
                                               const unsigned short* __restrict__ wvT,
                                               unsigned short* __restrict__ kws,
                                               unsigned short* __restrict__ vt) {
    const int b = blockIdx.x, ty = blockIdx.y;
    const int tid = threadIdx.x, w = tid >> 5, lane = tid & 31;
    const int lo = lane & 15, hi = lane >> 4;
    const int t0 = ty * 16;
    const float* ctxb = ctx + (size_t)b * NT_ * CD_;

    v8f acc[8];
#pragma unroll
    for (int i = 0; i < 8; ++i) acc[i] = v8f{0.f, 0.f, 0.f, 0.f, 0.f, 0.f, 0.f, 0.f};

    for (int kk = 0; kk < CD_ / 32; ++kk) {
        FragBF a = ldA_f32g(ctxb, NT_, t0, kk * 32, CD_, lane);  // shared by K and V
        FragBF bs[8];
#pragma unroll
        for (int i = 0; i < 8; ++i) {
            const int id = w * 8 + i;          // 0..63: 0..31 -> K, 32..63 -> V
            const unsigned short* wT = (id >> 5) ? wvT : wkT;
            bs[i] = ldB_bf(wT, (id & 31) * 16, kk * 32, CD_, lane);
        }
#pragma unroll
        for (int i = 0; i < 8; ++i)
            acc[i] = wmma_bf(a, bs[i], acc[i]);
    }
#pragma unroll
    for (int i = 0; i < 8; ++i) {
        const int id = w * 8 + i;
        const int mat = id >> 5;
        const int ct = id & 31;
        const int n = ct * 16 + lo;            // global inner dim
        const int h = n >> 6, d = n & 63;
#pragma unroll
        for (int e = 0; e < 8; ++e) {
            const int t = t0 + e + 8 * hi;
            unsigned short v = f2bf(acc[i][e]);
            if (mat == 0) kws[((size_t)(b * H_ + h) * NTP_ + t) * DH_ + d] = v;
            else          vt[((size_t)(b * H_ + h) * DH_ + d) * NTP_ + t] = v;
        }
    }
}

// ---------------------------------------------------------------------------
// Kernel 3: fused attention.  grid (NP/64, B), 256 threads = 8 wave32.
// Dynamic LDS layout (bytes), total exactly 160 KB -> 2 workgroups / WGP:
//   [0      , 65536)  ldsXO : ushort[64*512]  x tile (bf16), later reused for O
//   [65536  ,131072)  ldsQ  : ushort[64*512]  Q tile (scale folded in)
//   [131072 ,151552)  ldsS  : float [64*80]   raw scores
//   [151552 ,163840)  ldsP  : ushort[64*96]   softmax probs (bf16);
//                             cols 80..95 double as per-row reduction scratch
// ---------------------------------------------------------------------------
#define ATTN_SMEM 163840

__global__ __launch_bounds__(256) void attn_fused(const float* __restrict__ x,
                                                  const int* __restrict__ mask,
                                                  const unsigned short* __restrict__ wqT,
                                                  const unsigned short* __restrict__ woT,
                                                  const unsigned short* __restrict__ kws,
                                                  const unsigned short* __restrict__ vt,
                                                  const float* __restrict__ bo,
                                                  float* __restrict__ out) {
    extern __shared__ char smem[];
    unsigned short* ldsXO = (unsigned short*)smem;
    unsigned short* ldsQ  = (unsigned short*)(smem + 65536);
    float*          ldsS  = (float*)(smem + 131072);
    unsigned short* ldsP  = (unsigned short*)(smem + 151552);

    const int b = blockIdx.y;
    const int p0 = blockIdx.x * 64;
    const int tid = threadIdx.x, w = tid >> 5, lane = tid & 31;
    const int lo = lane & 15, hi = lane >> 4;
    const v8f vzero = {0.f, 0.f, 0.f, 0.f, 0.f, 0.f, 0.f, 0.f};

    // ---- Phase 1: stage x tile (64 x 512 fp32 -> bf16 LDS), coalesced float4 ----
    {
        const float* xb = x + ((size_t)b * NP_ + p0) * QD_;
        for (int i = tid; i < 64 * QD_ / 4; i += 256) {
            const int r = (i * 4) >> 9;       // /512
            const int c = (i * 4) & (QD_ - 1);
            float4 xv = *(const float4*)(xb + (size_t)r * QD_ + c);
            union { unsigned short u[4]; uint2 d; } pk;
            pk.u[0] = f2bf(xv.x); pk.u[1] = f2bf(xv.y);
            pk.u[2] = f2bf(xv.z); pk.u[3] = f2bf(xv.w);
            *(uint2*)&ldsXO[i * 4] = pk.d;
        }
    }
    __syncthreads();

    // ---- Phase 2: Q = X * Wq (64x512), scale 1/sqrt(64) folded in.
    //      Per kk: batch 1 global B-frag + 4 LDS A-frags, then 4 WMMAs. ----
    for (int ci = 0; ci < 4; ++ci) {
        const int ct = w + 8 * ci;            // 0..31 col tiles, unique per wave
        v8f acc[4] = {vzero, vzero, vzero, vzero};
#pragma unroll 4
        for (int kk = 0; kk < INNER_ / 32; ++kk) {
            FragBF bb = ldB_bf(wqT, ct * 16, kk * 32, QD_, lane);
            FragBF a[4];
#pragma unroll
            for (int rt = 0; rt < 4; ++rt)
                a[rt] = ldA_bf(ldsXO, rt * 16, kk * 32, QD_, lane);
#pragma unroll
            for (int rt = 0; rt < 4; ++rt)
                acc[rt] = wmma_bf(a[rt], bb, acc[rt]);
        }
        const int n = ct * 16 + lo;
#pragma unroll
        for (int rt = 0; rt < 4; ++rt)
#pragma unroll
            for (int e = 0; e < 8; ++e)
                ldsQ[(size_t)(rt * 16 + e + 8 * hi) * INNER_ + n] =
                    f2bf(acc[rt][e] * 0.125f);
    }
    __syncthreads();

    // ---- Phase 3: per-head attention ----
    for (int h = 0; h < H_; ++h) {
        // S = Q_h * K_h^T  (64 x 80), 20 tiles over 8 waves
        const unsigned short* kh = kws + (size_t)(b * H_ + h) * NTP_ * DH_;
#pragma unroll
        for (int i = 0; i < 3; ++i) {
            const int t = w + 8 * i;
            if (t < 20) {
                const int rt = t / 5, ct = t % 5;
                FragBF a[2], bb[2];
#pragma unroll
                for (int kk = 0; kk < 2; ++kk) {
                    a[kk]  = ldA_bf(ldsQ + h * DH_, rt * 16, kk * 32, INNER_, lane);
                    bb[kk] = ldB_bf(kh, ct * 16, kk * 32, DH_, lane);
                }
                v8f acc = vzero;
#pragma unroll
                for (int kk = 0; kk < 2; ++kk)
                    acc = wmma_bf(a[kk], bb[kk], acc);
                const int n = ct * 16 + lo;
#pragma unroll
                for (int e = 0; e < 8; ++e)
                    ldsS[(size_t)(rt * 16 + e + 8 * hi) * NTS_ + n] = acc[e];
            }
        }
        __syncthreads();

        // masked softmax: 4 threads per query row (strips of 24 over 96 cols).
        // All 4 strip-threads of a row sit in one wave32 -> scratch in P[80..95]
        // is safe (reads precede the zeroing stores in the wave's program order).
        {
            const int row = tid >> 2, q = tid & 3;
            float* srow = ldsS + (size_t)row * NTS_;
            unsigned short* prow = ldsP + (size_t)row * NTP_;
            float* rrow = (float*)(prow + 80);            // 8 f32 scratch per row
            const int* mrow = mask + ((size_t)b * NP_ + p0 + row) * NT_;
            const int tb = q * 24;
            const int te = (tb + 24 < NT_) ? tb + 24 : NT_;
            float mx = -3.0e38f;
            for (int t = tb; t < te; ++t) {
                float v = (mrow[t] == 0) ? -3.0e38f : srow[t];
                srow[t] = v;
                mx = fmaxf(mx, v);
            }
            rrow[q] = mx;
            __syncthreads();
            mx = fmaxf(fmaxf(rrow[0], rrow[1]), fmaxf(rrow[2], rrow[3]));
            float sum = 0.f;
            for (int t = tb; t < te; ++t) {
                float e_ = __expf(srow[t] - mx);
                srow[t] = e_;
                sum += e_;
            }
            rrow[4 + q] = sum;
            __syncthreads();
            sum = (rrow[4] + rrow[5]) + (rrow[6] + rrow[7]);
            const float inv = 1.0f / sum;
            for (int t = tb; t < tb + 24; ++t)
                prow[t] = (t < NT_) ? f2bf(srow[t] * inv) : (unsigned short)0;
        }
        __syncthreads();

        // O_h = P * V_h  (64 x 64): per wave one row-tile, two col tiles
        // sharing each A-fragment; overwrite the x buffer.
        const unsigned short* vh = vt + (size_t)(b * H_ + h) * DH_ * NTP_;
        {
            const int rt = w >> 1;
            const int ct0 = (w * 2) & 3;
            FragBF a[3], b0[3], b1[3];
#pragma unroll
            for (int kk = 0; kk < 3; ++kk) {
                a[kk]  = ldA_bf(ldsP, rt * 16, kk * 32, NTP_, lane);
                b0[kk] = ldB_bf(vh, ct0 * 16, kk * 32, NTP_, lane);
                b1[kk] = ldB_bf(vh, (ct0 + 1) * 16, kk * 32, NTP_, lane);
            }
            v8f o0 = vzero, o1 = vzero;
#pragma unroll
            for (int kk = 0; kk < 3; ++kk) {
                o0 = wmma_bf(a[kk], b0[kk], o0);
                o1 = wmma_bf(a[kk], b1[kk], o1);
            }
            const int n0 = h * DH_ + ct0 * 16 + lo;
#pragma unroll
            for (int e = 0; e < 8; ++e) {
                const size_t r = (size_t)(rt * 16 + e + 8 * hi) * INNER_;
                ldsXO[r + n0]      = f2bf(o0[e]);
                ldsXO[r + n0 + 16] = f2bf(o1[e]);
            }
        }
        __syncthreads();
    }

    // ---- Phase 4: out = O * Wo + bo  (64 x 512 fp32), batched loads per kk ----
    for (int ci = 0; ci < 4; ++ci) {
        const int ct = w + 8 * ci;
        const int n = ct * 16 + lo;
        const float bias = bo[n];
        v8f acc[4] = {vzero, vzero, vzero, vzero};
#pragma unroll 4
        for (int kk = 0; kk < INNER_ / 32; ++kk) {
            FragBF bb = ldB_bf(woT, ct * 16, kk * 32, INNER_, lane);
            FragBF a[4];
#pragma unroll
            for (int rt = 0; rt < 4; ++rt)
                a[rt] = ldA_bf(ldsXO, rt * 16, kk * 32, INNER_, lane);
#pragma unroll
            for (int rt = 0; rt < 4; ++rt)
                acc[rt] = wmma_bf(a[rt], bb, acc[rt]);
        }
#pragma unroll
        for (int rt = 0; rt < 4; ++rt)
#pragma unroll
            for (int e = 0; e < 8; ++e) {
                const int row = p0 + rt * 16 + e + 8 * hi;
                out[((size_t)b * NP_ + row) * QD_ + n] = acc[rt][e] + bias;
            }
    }
}

// ---------------------------------------------------------------------------
// Launch
// ---------------------------------------------------------------------------
extern "C" void kernel_launch(void* const* d_in, const int* in_sizes, int n_in,
                              void* d_out, int out_size, void* d_ws, size_t ws_size,
                              hipStream_t stream) {
    (void)in_sizes; (void)n_in; (void)out_size; (void)ws_size;
    const float* x       = (const float*)d_in[0];
    const float* context = (const float*)d_in[1];
    const int*   mask    = (const int*)d_in[2];
    const float* Wq      = (const float*)d_in[3];
    const float* Wk      = (const float*)d_in[4];
    const float* Wv      = (const float*)d_in[5];
    const float* Wo      = (const float*)d_in[6];
    const float* bo      = (const float*)d_in[7];
    float* out = (float*)d_out;

    unsigned short* ws = (unsigned short*)d_ws;
    unsigned short* wqT = ws;                       // 512*512
    unsigned short* wkT = ws + 262144;              // 768*512
    unsigned short* wvT = ws + 655360;              // 768*512
    unsigned short* woT = ws + 1048576;             // 512*512
    unsigned short* kws = ws + 1310720;             // 16*8*96*64
    unsigned short* vtw = ws + 2097152;             // 16*8*64*96

    transpose_w<<<(512 * 512 + 255) / 256, 256, 0, stream>>>(Wq, wqT, 512, 512);
    transpose_w<<<(768 * 512 + 255) / 256, 256, 0, stream>>>(Wk, wkT, 768, 512);
    transpose_w<<<(768 * 512 + 255) / 256, 256, 0, stream>>>(Wv, wvT, 768, 512);
    transpose_w<<<(512 * 512 + 255) / 256, 256, 0, stream>>>(Wo, woT, 512, 512);

    kv_proj<<<dim3(B_, NTP_ / 16), 256, 0, stream>>>(context, wkT, wvT, kws, vtw);

    attn_fused<<<dim3(NP_ / 64, B_), 256, ATTN_SMEM, stream>>>(
        x, mask, wqT, woT, kws, vtw, bo, out);
}
